// Matcher_10771777978480
// MI455X (gfx1250) — matrix operand order repaired and checked by reading, not compiled, not run on hardware
//
#include <hip/hip_runtime.h>
#include <math.h>

// ---------------- types for WMMA ----------------
typedef __bf16 bf16;
typedef bf16  v16bf __attribute__((ext_vector_type(16)));
typedef float v8f   __attribute__((ext_vector_type(8)));
typedef unsigned int ui4 __attribute__((ext_vector_type(4)));

#define D_DIM   768          // feature dim (reference fixed)
#define D4      (D_DIM/4)    // float4 per row
#define GSTRIDE 776          // padded bf16 row stride in LDS (768+8) -> conflict-free ds_load_b128
#define WBSTRIDE 65          // staging row stride (floats), odd -> distinct banks per query
#define KTOP    10
#define NEGINF  (-3.402823466e38f)

// ---- ordering identical to jax.lax.top_k: higher sim wins, lower index on ties ----
__device__ __forceinline__ bool better(float s1, int i1, float s2, int i2) {
    return (s1 > s2) || (s1 == s2 && i1 < i2);
}

__device__ __forceinline__ void ins10(float (&ts)[KTOP], int (&ti)[KTOP], float v, int idx) {
    if (!better(v, idx, ts[KTOP-1], ti[KTOP-1])) return;
    ts[KTOP-1] = v; ti[KTOP-1] = idx;
#pragma unroll
    for (int k = KTOP-1; k > 0; --k) {
        if (better(ts[k], ti[k], ts[k-1], ti[k-1])) {
            float tv = ts[k]; ts[k] = ts[k-1]; ts[k-1] = tv;
            int   tj = ti[k]; ti[k] = ti[k-1]; ti[k-1] = tj;
        }
    }
}

__device__ __forceinline__ void ins10p(float* ts, int* ti, float v, int idx) {
    if (!better(v, idx, ts[KTOP-1], ti[KTOP-1])) return;
    ts[KTOP-1] = v; ti[KTOP-1] = idx;
    for (int k = KTOP-1; k > 0; --k) {
        if (better(ts[k], ti[k], ts[k-1], ti[k-1])) {
            float tv = ts[k]; ts[k] = ts[k-1]; ts[k-1] = tv;
            int   tj = ti[k]; ti[k] = ti[k-1]; ti[k-1] = tj;
        } else break;
    }
}

// A fragment (16x32 bf16, M=lane%16): K = h*16 + {0..15}, contiguous 32B
__device__ __forceinline__ v16bf ldA(const bf16* p) {
    union { ui4 q[2]; v16bf v; } u;
    u.q[0] = *(const ui4*)(p);
    u.q[1] = *(const ui4*)(p + 8);
    return u.v;
}
// B fragment (32x16 bf16, N=lane%16): K = h*16 + {0..15}, contiguous 32B
__device__ __forceinline__ v16bf ldB(const bf16* p) {
    union { ui4 q[2]; v16bf v; } u;
    u.q[0] = *(const ui4*)(p);
    u.q[1] = *(const ui4*)(p + 8);
    return u.v;
}

// ---------------- kernel 1: normalize q, split fp32 -> bf16 hi/lo ----------------
__global__ void __launch_bounds__(256) qprep_kernel(const float* __restrict__ q,
                                                    bf16* __restrict__ qhi,
                                                    bf16* __restrict__ qlo) {
    __shared__ float red[256];
    const int b = blockIdx.x;
    const float* row = q + (size_t)b * D_DIM;
    float s = 0.f;
    for (int e = threadIdx.x; e < D_DIM; e += 256) { float v = row[e]; s += v * v; }
    red[threadIdx.x] = s; __syncthreads();
    for (int off = 128; off > 0; off >>= 1) {
        if (threadIdx.x < off) red[threadIdx.x] += red[threadIdx.x + off];
        __syncthreads();
    }
    const float scale = 1.0f / fmaxf(sqrtf(red[0]), 1e-12f);
    for (int e = threadIdx.x; e < D_DIM; e += 256) {
        float xn = row[e] * scale;
        bf16 h = (bf16)xn;
        bf16 l = (bf16)(xn - (float)h);
        qhi[(size_t)b * D_DIM + e] = h;
        qlo[(size_t)b * D_DIM + e] = l;
    }
}

// ---------------- kernel 2: fused bf16x3 WMMA GEMM + per-tile top-10 ----------------
// Block = 256 threads (8 waves). Block owns 256 gallery rows = 4 subtiles of 64.
// Wave w handles query chunks {w, w+8, w+16, w+24} of 16 queries (B must be 512).
__global__ void __launch_bounds__(256) gemm_topk_kernel(
    const float* __restrict__ g,
    const bf16*  __restrict__ qhi, const bf16* __restrict__ qlo,
    const int* __restrict__ gpids, const int* __restrict__ gcams,
    const int* __restrict__ qpids, const int* __restrict__ qcams,
    float* __restrict__ candS, int* __restrict__ candI,
    int M, int B)
{
    extern __shared__ char smem[];
    bf16*  g_hi    = (bf16*)smem;                                    // 64*GSTRIDE
    bf16*  g_lo    = g_hi + 64 * GSTRIDE;                            // 64*GSTRIDE
    float* scale_s = (float*)(smem + (size_t)2 * 64 * GSTRIDE * sizeof(bf16)); // 64
    int*   gpid_s  = (int*)(scale_s + 64);                           // 64
    int*   gcam_s  = gpid_s + 64;                                    // 64
    float* wavebuf = (float*)(gcam_s + 64);                          // 8*16*WBSTRIDE
    float* pairS   = wavebuf + 8 * 16 * WBSTRIDE;                    // 8*32*KTOP
    int*   pairI   = (int*)(pairS + 8 * 32 * KTOP);                  // 8*32*KTOP
    float* topS    = (float*)(pairI + 8 * 32 * KTOP);                // B*KTOP
    int*   topI    = (int*)(topS + (size_t)B * KTOP);                // B*KTOP

    const int tid  = threadIdx.x;
    const int lane = tid & 31;
    const int wave = tid >> 5;
    const int nl   = lane & 15;     // query column within chunk / A row within subtile
    const int h    = lane >> 4;     // lane-half selector
    const int block_base = blockIdx.x * 256;

    // init persistent per-query top lists
    for (int i = tid; i < B * KTOP; i += 256) { topS[i] = NEGINF; topI[i] = 0x7fffffff; }

    const int nchunks = B >> 4;     // 32 when B=512 (multiple of 8 -> uniform barriers)

    for (int s = 0; s < 4; ++s) {
        const int mbase = block_base + s * 64;
        __syncthreads();   // previous tile fully consumed
        // --- load 64x768 fp32 tile, convert to bf16 hi/lo in LDS (coalesced float4) ---
        for (int i = tid; i < 64 * D4; i += 256) {
            int r = i / D4, c4 = i - r * D4;
            int gm = mbase + r;
            float4 v = {0.f, 0.f, 0.f, 0.f};
            if (gm < M) v = ((const float4*)g)[(size_t)gm * D4 + c4];
            int col = c4 * 4;
            bf16* ph = g_hi + r * GSTRIDE + col;
            bf16* pl = g_lo + r * GSTRIDE + col;
            float x0 = v.x, x1 = v.y, x2 = v.z, x3 = v.w;
            bf16 h0 = (bf16)x0, h1 = (bf16)x1, h2 = (bf16)x2, h3 = (bf16)x3;
            ph[0] = h0; ph[1] = h1; ph[2] = h2; ph[3] = h3;
            pl[0] = (bf16)(x0 - (float)h0); pl[1] = (bf16)(x1 - (float)h1);
            pl[2] = (bf16)(x2 - (float)h2); pl[3] = (bf16)(x3 - (float)h3);
        }
        __syncthreads();
        // --- row inverse norms (normalize-after-dot == normalize-before-dot) ---
        for (int j = 0; j < 8; ++j) {
            int r = wave + j * 8;
            float sum = 0.f;
            for (int c = lane; c < D_DIM; c += 32) {
                float v = (float)g_hi[r * GSTRIDE + c] + (float)g_lo[r * GSTRIDE + c];
                sum += v * v;
            }
            for (int o = 16; o > 0; o >>= 1) sum += __shfl_down(sum, o, 32);
            if (lane == 0) scale_s[r] = 1.0f / fmaxf(sqrtf(sum), 1e-12f);
        }
        if (tid < 64) {
            int gm = mbase + tid;
            gpid_s[tid] = (gm < M) ? gpids[gm] : -1;
            gcam_s[tid] = (gm < M) ? gcams[gm] : -1;
        }
        __syncthreads();

        // --- compute: each wave does 4 chunks of 16 queries over the 64-row subtile ---
        for (int qc = wave; qc < nchunks; qc += 8) {
            const int n = qc * 16 + nl;                       // global query index
            const bf16* qrh = qhi + (size_t)n * D_DIM + h * 16;
            const bf16* qrl = qlo + (size_t)n * D_DIM + h * 16;
            const bf16* arh = g_hi + nl * GSTRIDE + h * 16;
            const bf16* arl = g_lo + nl * GSTRIDE + h * 16;

            v8f acc[4] = {};
            for (int kb = 0; kb < D_DIM; kb += 32) {
                v16bf bh = ldB(qrh + kb);
                v16bf bl = ldB(qrl + kb);
#pragma unroll
                for (int t = 0; t < 4; ++t) {
                    v16bf ah = ldA(arh + t * 16 * GSTRIDE + kb);
                    v16bf al = ldA(arl + t * 16 * GSTRIDE + kb);
                    // bf16x3 fp32 emulation: hi*hi + lo*hi + hi*lo
                    acc[t] = __builtin_amdgcn_wmma_f32_16x16x32_bf16(false, ah, false, bh, (short)0, acc[t], false, false);
                    acc[t] = __builtin_amdgcn_wmma_f32_16x16x32_bf16(false, al, false, bh, (short)0, acc[t], false, false);
                    acc[t] = __builtin_amdgcn_wmma_f32_16x16x32_bf16(false, ah, false, bl, (short)0, acc[t], false, false);
                }
            }

            // stage raw dots to wave-private LDS: wavebuf[wave][n_local][m_local]
            float* wb = wavebuf + (wave * 16 + nl) * WBSTRIDE;
#pragma unroll
            for (int t = 0; t < 4; ++t)
#pragma unroll
                for (int r = 0; r < 8; ++r)
                    wb[t * 16 + r + h * 8] = acc[t][r];

            // scan 32 values per lane (lane-half picks m range), mask + scale, local top-10
            const int mypid = qpids[n], mycam = qcams[n];
            float t10s[KTOP]; int t10i[KTOP];
#pragma unroll
            for (int k = 0; k < KTOP; ++k) { t10s[k] = NEGINF; t10i[k] = 0x7fffffff; }
            for (int j = 0; j < 32; ++j) {
                int ml = h * 32 + j;
                int gm = mbase + ml;
                if (gm < M) {
                    float v = wb[ml] * scale_s[ml];
                    if (gpid_s[ml] == mypid && gcam_s[ml] == mycam) v = -1.0f;
                    ins10(t10s, t10i, v, gm);
                }
            }
            // publish lane-local lists, then lane-half 0 merges both halves into
            // the persistent per-query list (single writer per query -> no atomics;
            // same-wave DS ops are in-order).
            float* pS = pairS + (wave * 32 + lane) * KTOP;
            int*   pI = pairI + (wave * 32 + lane) * KTOP;
#pragma unroll
            for (int k = 0; k < KTOP; ++k) { pS[k] = t10s[k]; pI[k] = t10i[k]; }
            if (h == 0) {
                float* dS = topS + (size_t)n * KTOP;
                int*   dI = topI + (size_t)n * KTOP;
                for (int src = 0; src < 2; ++src) {
                    const float* sS = pairS + (wave * 32 + nl + src * 16) * KTOP;
                    const int*   sI = pairI + (wave * 32 + nl + src * 16) * KTOP;
                    for (int k = 0; k < KTOP; ++k) {
                        if (!better(sS[k], sI[k], dS[KTOP-1], dI[KTOP-1])) break; // lists sorted
                        ins10p(dS, dI, sS[k], sI[k]);
                    }
                }
            }
        }
    }
    __syncthreads();
    // write this block's per-query candidates: layout [tile][query][10]
    for (int i = tid; i < B * KTOP; i += 256) {
        candS[(size_t)blockIdx.x * B * KTOP + i] = topS[i];
        candI[(size_t)blockIdx.x * B * KTOP + i] = topI[i];
    }
}

// ---------------- kernel 3: merge candidates, gather, write 5 outputs ----------------
__global__ void __launch_bounds__(32) finalize_kernel(
    const float* __restrict__ candS, const int* __restrict__ candI,
    const int* __restrict__ qpids, const int* __restrict__ gpids,
    const int* __restrict__ gcams,
    float* __restrict__ out, int ntiles, int B, int M)
{
    __shared__ float mS[32 * KTOP];
    __shared__ int   mI[32 * KTOP];
    const int b = blockIdx.x, lane = threadIdx.x;
    const int NC = ntiles * KTOP;

    float t10s[KTOP]; int t10i[KTOP];
#pragma unroll
    for (int k = 0; k < KTOP; ++k) { t10s[k] = NEGINF; t10i[k] = 0x7fffffff; }
    for (int i = lane; i < NC; i += 32) {
        int t = i / KTOP, k = i - t * KTOP;
        size_t off = ((size_t)t * B + b) * KTOP + k;
        int ii = candI[off];
        if (ii != 0x7fffffff) ins10(t10s, t10i, candS[off], ii);
    }
#pragma unroll
    for (int k = 0; k < KTOP; ++k) { mS[lane * KTOP + k] = t10s[k]; mI[lane * KTOP + k] = t10i[k]; }
    __syncthreads();
    if (lane == 0) {
        float fs[KTOP]; int fi[KTOP];
#pragma unroll
        for (int k = 0; k < KTOP; ++k) { fs[k] = NEGINF; fi[k] = 0x7fffffff; }
        for (int j = 0; j < 32 * KTOP; ++j)
            if (mI[j] != 0x7fffffff) ins10(fs, fi, mS[j], mI[j]);
#pragma unroll
        for (int k = 0; k < KTOP; ++k) { mS[k] = fs[k]; mI[k] = fi[k]; }
    }
    __syncthreads();
    if (lane < KTOP) {
        const int k   = lane;
        const float sim = mS[k];
        const int   idx = mI[k];
        float pid = -1.f, cam = -1.f;
        if ((unsigned)idx < (unsigned)M) { pid = (float)gpids[idx]; cam = (float)gcams[idx]; }
        const float match = ((float)qpids[b] == pid) ? 1.0f : 0.0f;
        const size_t BK = (size_t)B * KTOP;
        out[0 * BK + (size_t)b * KTOP + k] = match;       // matches
        out[1 * BK + (size_t)b * KTOP + k] = sim;         // topk_sim
        out[2 * BK + (size_t)b * KTOP + k] = pid;         // match_pids
        out[3 * BK + (size_t)b * KTOP + k] = cam;         // match_camids
        out[4 * BK + (size_t)b * KTOP + k] = (float)idx;  // topk_index
    }
}

// ---------------- host launcher ----------------
extern "C" void kernel_launch(void* const* d_in, const int* in_sizes, int n_in,
                              void* d_out, int out_size, void* d_ws, size_t ws_size,
                              hipStream_t stream) {
    const float* q     = (const float*)d_in[0];
    const float* g     = (const float*)d_in[1];
    const int*   qpids = (const int*)d_in[2];
    const int*   qcams = (const int*)d_in[3];
    const int*   gpids = (const int*)d_in[4];
    const int*   gcams = (const int*)d_in[5];

    const int B = in_sizes[2];                  // 512
    const int M = in_sizes[4];                  // 100000
    const int ntiles = (M + 255) / 256;         // 391

    // workspace layout
    bf16*  qhi   = (bf16*)d_ws;
    bf16*  qlo   = qhi + (size_t)B * D_DIM;
    float* candS = (float*)(qlo + (size_t)B * D_DIM);
    int*   candI = (int*)(candS + (size_t)ntiles * B * KTOP);

    qprep_kernel<<<B, 256, 0, stream>>>(q, qhi, qlo);

    const size_t lds = (size_t)2 * 64 * GSTRIDE * sizeof(bf16)   // g tile hi+lo
                     + 3 * 64 * 4                                // scale/pid/cam
                     + 8 * 16 * WBSTRIDE * 4                     // staging
                     + 2 * 8 * 32 * KTOP * 4                     // pair lists
                     + 2 * (size_t)B * KTOP * 4;                 // persistent tops
    gemm_topk_kernel<<<ntiles, 256, lds, stream>>>(g, qhi, qlo, gpids, gcams,
                                                   qpids, qcams, candS, candI, M, B);

    finalize_kernel<<<B, 32, 0, stream>>>(candS, candI, qpids, gpids, gcams,
                                          (float*)d_out, ntiles, B, M);
}
